// E24Cell_29575144800928
// MI455X (gfx1250) — matrix operand order repaired and checked by reading, not compiled, not run on hardware
//
#include <hip/hip_runtime.h>
#include <hip/hip_bf16.h>
#include <math.h>

#define B_  16
#define T_  256
#define D_  1024
#define N_  64
#define K2  2048            // 2*D
#define AP  2056            // padded LDS row stride (elements) -> no 16-way bank conflicts

typedef __bf16    v16bf __attribute__((ext_vector_type(16)));
typedef float     v8f   __attribute__((ext_vector_type(8)));
typedef unsigned  u32x4 __attribute__((ext_vector_type(4)));

union FragBF16 { u32x4 q[2]; v16bf v; };

__device__ __forceinline__ unsigned short f32_to_bf16_rne(float f) {
    union { float f; unsigned u; } x; x.f = f;
    unsigned r = x.u + 0x7FFFu + ((x.u >> 16) & 1u);   // round-to-nearest-even
    return (unsigned short)(r >> 16);
}

__device__ __forceinline__ float wave_sum(float v) {
#pragma unroll
    for (int off = 16; off > 0; off >>= 1) v += __shfl_xor(v, off, 32);
    return v;
}

// Sense-reversing grid barrier (16 blocks, all co-resident by construction).
__device__ __forceinline__ void grid_sync(unsigned* bar, unsigned nblocks) {
    __syncthreads();
    if (threadIdx.x == 0) {
        __threadfence();
        unsigned gen  = __hip_atomic_load(&bar[1], __ATOMIC_ACQUIRE, __HIP_MEMORY_SCOPE_AGENT);
        unsigned prev = __hip_atomic_fetch_add(&bar[0], 1u, __ATOMIC_ACQ_REL, __HIP_MEMORY_SCOPE_AGENT);
        if (prev == nblocks - 1u) {
            __hip_atomic_store(&bar[0], 0u, __ATOMIC_RELAXED, __HIP_MEMORY_SCOPE_AGENT);
            __hip_atomic_fetch_add(&bar[1], 1u, __ATOMIC_ACQ_REL, __HIP_MEMORY_SCOPE_AGENT);
        } else {
            while (__hip_atomic_load(&bar[1], __ATOMIC_ACQUIRE, __HIP_MEMORY_SCOPE_AGENT) == gen)
                __builtin_amdgcn_s_sleep(2);
        }
        __threadfence();
    }
    __syncthreads();
}

// Prologue: W fp32 -> bf16 (L2-resident reuse, halves per-step L2 traffic,
// 8x denser WMMA than f32 16x16x4), copy mutable state, reset barrier.
__global__ void e24_prologue(const float* __restrict__ W,
                             const float* __restrict__ tape_in,
                             const float* __restrict__ hw_in,
                             unsigned short* __restrict__ Wb,
                             float* __restrict__ tape,
                             float* __restrict__ hwbuf,
                             unsigned* __restrict__ bar) {
    const size_t stride = (size_t)gridDim.x * blockDim.x;
    size_t i = (size_t)blockIdx.x * blockDim.x + threadIdx.x;
    for (size_t j = i; j < (size_t)K2 * K2;       j += stride) Wb[j]    = f32_to_bf16_rne(W[j]);
    for (size_t j = i; j < (size_t)B_ * N_ * D_;  j += stride) tape[j]  = tape_in[j];
    for (size_t j = i; j < (size_t)B_ * D_;       j += stride) hwbuf[j] = hw_in[j];
    if (i == 0) { bar[0] = 0u; bar[1] = 0u; }
}

// Persistent kernel: 16 blocks x 256 threads (8 wave32 / block = 128 waves).
// Per step: stage A=[h_w|x_t] bf16 in LDS -> 128 N-tiles of WMMA GEMM
// -> grid barrier -> per-batch attention/tanh/tape update -> grid barrier.
__global__ __launch_bounds__(256)
void e24_persistent(const float* __restrict__ x_seq,
                    const float* __restrict__ b_h,
                    const unsigned short* __restrict__ Wb,
                    float* __restrict__ out_ws,     // 16 x 2048 f32 step output
                    float* __restrict__ tape,       // B x N x D state
                    float* __restrict__ hw,         // B x D state
                    float* __restrict__ out_all,    // B x T x D
                    float* __restrict__ out_tape,   // B x N x D
                    float* __restrict__ out_hw,     // B x D
                    unsigned* __restrict__ bar)
{
    __shared__ union SM {
        unsigned short a16[16 * AP];                // GEMM A staging (bf16)
        struct {                                    // attention phase scratch
            float hw[D_]; float hnew[D_];
            float sc[N_]; float attn[N_]; float wsc[N_]; float wattn[N_];
        } c;
    } sm;

    const int b    = blockIdx.x;                    // batch owned in phase C
    const int tid  = threadIdx.x;
    const int wv   = tid >> 5;
    const int lane = tid & 31;
    const int m16  = lane & 15;                     // M (A row / batch) and N (D col) lane index
    const int half = lane >> 4;                     // upper half-wave: +8 / +16 K groups
    const int j0   = (b * 8 + wv) * 16;             // this wave's 16 output columns
    const float scale = 0.03125f;                   // 1/sqrt(1024)

    for (int t = 0; t < T_; ++t) {
        // ---- Phase A: stage A = [h_w | x_t] as bf16 into LDS (padded stride) ----
        for (int i = tid; i < 16 * K2; i += 256) {
            int row = i >> 11, k = i & (K2 - 1);
            float v = (k < D_) ? hw[row * D_ + k]
                               : x_seq[((size_t)row * T_ + t) * D_ + (k - D_)];
            sm.a16[row * AP + k] = f32_to_bf16_rne(v);
        }
        __syncthreads();

        // ---- Phase B: out[16 x 2048] = A @ W^T, v_wmma_f32_16x16x32_bf16 ----
        {
            v8f acc = {0.f,0.f,0.f,0.f,0.f,0.f,0.f,0.f};
            // A lane layout: row M=m16; elems 0..7 -> K=k0+half*8.., elems 8..15 -> K=k0+16+half*8..
            const unsigned short* aBase = &sm.a16[m16 * AP + half * 8];
            // B lane layout: col N=m16 -> W row j0+m16; elems 0..7 -> K=k0+half*8.., elems 8..15 -> K=k0+16+half*8..
            const unsigned short* wBase = Wb + (size_t)(j0 + m16) * K2 + half * 8;
#pragma unroll 4
            for (int k0 = 0; k0 < K2; k0 += 32) {
                FragBF16 a, w;
                a.q[0] = *(const u32x4*)(aBase + k0);
                a.q[1] = *(const u32x4*)(aBase + k0 + 16);
                w.q[0] = *(const u32x4*)(wBase + k0);
                w.q[1] = *(const u32x4*)(wBase + k0 + 16);
                __builtin_prefetch(wBase + k0 + 512, 0, 1);
                acc = __builtin_amdgcn_wmma_f32_16x16x32_bf16(
                          false, a.v, false, w.v, (short)0, acc, false, false);
            }
            // D layout: VGPR r -> M = half*8 + r, N = m16
#pragma unroll
            for (int r = 0; r < 8; ++r)
                out_ws[(size_t)(half * 8 + r) * K2 + j0 + m16] = acc[r];
        }
        grid_sync(bar, B_);

        // ---- Phase C: attention + state update, block b owns batch b ----
        for (int i = tid; i < D_; i += 256) sm.c.hw[i] = hw[b * D_ + i];
        __syncthreads();

        // read-attention scores: wave wv handles slots n = wv*8 .. wv*8+7
        for (int s = 0; s < 8; ++s) {
            int n = wv * 8 + s;
            const float* trow = tape + ((size_t)(b * N_ + n)) * D_;
            float p = 0.f;
            for (int d = lane; d < D_; d += 32) p += trow[d] * sm.c.hw[d];
            p = wave_sum(p);
            if (lane == 0) sm.c.sc[n] = p * scale;
        }
        __syncthreads();
        {   // stable softmax (redundant per-thread reduction over 64 LDS values)
            float mx = -__builtin_inff();
            for (int n = 0; n < N_; ++n) mx = fmaxf(mx, sm.c.sc[n]);
            float den = 0.f;
            for (int n = 0; n < N_; ++n) den += __expf(sm.c.sc[n] - mx);
            float inv = 1.f / den;
            if (tid < N_) sm.c.attn[tid] = __expf(sm.c.sc[tid] - mx) * inv;
        }
        __syncthreads();

        // read vector + h_w_new = tanh(h_update + read + b_h)
        for (int d = tid; d < D_; d += 256) {
            float r = 0.f;
            for (int n = 0; n < N_; ++n)
                r += sm.c.attn[n] * tape[((size_t)(b * N_ + n)) * D_ + d];
            float hval = tanhf(out_ws[b * K2 + d] + r + b_h[d]);
            sm.c.hnew[d] = hval;
            hw[b * D_ + d] = hval;
            out_all[((size_t)b * T_ + t) * D_ + d] = hval;
        }
        __syncthreads();

        // write-attention scores with h_w_new
        for (int s = 0; s < 8; ++s) {
            int n = wv * 8 + s;
            const float* trow = tape + ((size_t)(b * N_ + n)) * D_;
            float p = 0.f;
            for (int d = lane; d < D_; d += 32) p += trow[d] * sm.c.hnew[d];
            p = wave_sum(p);
            if (lane == 0) sm.c.wsc[n] = p * scale;
        }
        __syncthreads();
        {
            float mx = -__builtin_inff();
            for (int n = 0; n < N_; ++n) mx = fmaxf(mx, sm.c.wsc[n]);
            float den = 0.f;
            for (int n = 0; n < N_; ++n) den += __expf(sm.c.wsc[n] - mx);
            float inv = 1.f / den;
            if (tid < N_) sm.c.wattn[tid] = __expf(sm.c.wsc[tid] - mx) * inv;
        }
        __syncthreads();

        // tape RMW: h_tape = (1-wa)*h_tape + wa*write_val
        for (int i = tid; i < N_ * D_; i += 256) {
            int n = i >> 10, d = i & (D_ - 1);
            float wa = sm.c.wattn[n];
            size_t idx = (size_t)b * N_ * D_ + i;
            tape[idx] = (1.f - wa) * tape[idx] + wa * out_ws[b * K2 + D_ + d];
        }
        grid_sync(bar, B_);
    }

    // epilogue: final tape + h_work
    for (int i = tid; i < N_ * D_; i += 256)
        out_tape[(size_t)b * N_ * D_ + i] = tape[(size_t)b * N_ * D_ + i];
    for (int i = tid; i < D_; i += 256)
        out_hw[b * D_ + i] = hw[b * D_ + i];
}

extern "C" void kernel_launch(void* const* d_in, const int* in_sizes, int n_in,
                              void* d_out, int out_size, void* d_ws, size_t ws_size,
                              hipStream_t stream) {
    (void)in_sizes; (void)n_in; (void)out_size; (void)ws_size;
    const float* x_seq  = (const float*)d_in[0];
    const float* h_tape = (const float*)d_in[1];
    const float* h_work = (const float*)d_in[2];
    const float* W_all  = (const float*)d_in[3];
    const float* b_h    = (const float*)d_in[4];
    float* out = (float*)d_out;

    char* ws = (char*)d_ws;
    unsigned short* Wb = (unsigned short*)(ws);                               // 8 MB bf16 weights
    float* out_ws = (float*)(ws + 8388608);                                   // 128 KB step GEMM out
    float* tape   = (float*)(ws + 8388608 + 131072);                          // 4 MB tape state
    float* hwbuf  = (float*)(ws + 8388608 + 131072 + 4194304);                // 64 KB h_w state
    unsigned* bar = (unsigned*)(ws + 8388608 + 131072 + 4194304 + 65536);     // barrier

    float* out_all  = out;                                                    // B*T*D
    float* out_tape = out + (size_t)B_ * T_ * D_;                             // B*N*D
    float* out_hw   = out_tape + (size_t)B_ * N_ * D_;                        // B*D

    e24_prologue<<<256, 256, 0, stream>>>(W_all, h_tape, h_work, Wb, tape, hwbuf, bar);
    e24_persistent<<<B_, 256, 0, stream>>>(x_seq, b_h, Wb, out_ws, tape, hwbuf,
                                           out_all, out_tape, out_hw, bar);
}
